// BahdanauAttention_65584150610573
// MI455X (gfx1250) — compile-verified, matrix operand order checked by reading, not run on hardware
//
#include <hip/hip_runtime.h>
#include <hip/hip_bf16.h>
#include <math.h>

typedef __attribute__((ext_vector_type(16))) __bf16 v16bf;
typedef __attribute__((ext_vector_type(8)))  float  v8f;

#define B_    32
#define S_    2048
#define ENC2_ 1024
#define ATT_  1024
#define ROWS  64      // S-rows staged per workgroup (4 waves x 16 rows)
#define KTILE 32      // K per bf16 WMMA

__device__ __forceinline__ unsigned short f32_bf16_rne(float f) {
    unsigned int u = __float_as_uint(f);
    u += 0x7FFFu + ((u >> 16) & 1u);
    return (unsigned short)(u >> 16);
}
__device__ __forceinline__ float bf16_as_f32(unsigned short h) {
    return __uint_as_float(((unsigned int)h) << 16);
}

// ---------------------------------------------------------------------------
// One-time: transpose W_enc (K x N) -> N-major bf16 hi/lo planes in workspace
// so WMMA B-fragments are contiguous 16B loads (L2-resident, 4MB total).
// ---------------------------------------------------------------------------
__global__ void wprep_kernel(const float* __restrict__ Wenc,
                             unsigned short* __restrict__ whi,
                             unsigned short* __restrict__ wlo) {
    size_t gid = (size_t)blockIdx.x * blockDim.x + threadIdx.x;
    if (gid >= (size_t)ENC2_ * ATT_) return;
    int k = (int)(gid >> 10);          // row of W_enc
    int n = (int)(gid & 1023);         // col of W_enc
    float v = Wenc[gid];
    unsigned short hi = f32_bf16_rne(v);
    unsigned short lo = f32_bf16_rne(v - bf16_as_f32(hi));
    size_t o = (size_t)n * ENC2_ + k;  // transposed: column-K contiguous
    whi[o] = hi;
    wlo[o] = lo;
}

// ---------------------------------------------------------------------------
// decoder_proj[b][n] = decoder_hidden[b] . W_dec[:,n] + b_dec[n] + b_enc[n]
// ---------------------------------------------------------------------------
__global__ __launch_bounds__(256) void dproj_kernel(
    const float* __restrict__ dh, const float* __restrict__ Wdec,
    const float* __restrict__ bdec, const float* __restrict__ benc,
    float* __restrict__ dproj) {
    __shared__ float h[ENC2_];
    const int b = blockIdx.x;
    for (int i = threadIdx.x; i < ENC2_; i += 256) h[i] = dh[(size_t)b * ENC2_ + i];
    __syncthreads();
#pragma unroll
    for (int jj = 0; jj < 4; ++jj) {
        int n = threadIdx.x + jj * 256;
        float acc = 0.f;
        for (int k = 0; k < ENC2_; ++k)
            acc = fmaf(h[k], Wdec[(size_t)k * ATT_ + n], acc);
        dproj[(size_t)b * ATT_ + n] = acc + bdec[n] + benc[n];
    }
}

// ---------------------------------------------------------------------------
// Main kernel: energy[b][s] = tanh(enc[b,s,:]@W_enc + dproj[b,:]) . w_e + b_e
// bf16x3 split GEMM on v_wmma_f32_16x16x32_bf16, fp32 accumulate.
// One WG = 4 waves = 64 S-rows; full K staged in 256KB LDS (read HBM once).
// 256KB LDS => 1 wave/SIMD, so ILP comes from 2 N-columns per iteration and
// independent accumulator chains (c_hh / c_lo per column).
// ---------------------------------------------------------------------------
__global__ __launch_bounds__(128) void energy_kernel(
    const float* __restrict__ enc,
    const unsigned short* __restrict__ whi,
    const unsigned short* __restrict__ wlo,
    const float* __restrict__ dproj,
    const float* __restrict__ w_e,
    const float* __restrict__ b_e_p,
    const unsigned char* __restrict__ mask,
    float* __restrict__ energy) {

    __shared__ __align__(16) unsigned short a_hi[ROWS * ENC2_];
    __shared__ __align__(16) unsigned short a_lo[ROWS * ENC2_];

    const int b  = blockIdx.x >> 5;          // S_/ROWS == 32 tiles per batch
    const int s0 = (blockIdx.x & 31) * ROWS;

    // ---- stage A tile: fp32 -> bf16 hi/lo planes (float4 in, uint2 out) ----
    const float4* src = (const float4*)(enc + ((size_t)b * S_ + s0) * ENC2_);
    uint2* dsth = (uint2*)a_hi;
    uint2* dstl = (uint2*)a_lo;
    for (int i = threadIdx.x; i < ROWS * ENC2_ / 4; i += 128) {
        float4 v = src[i];
        unsigned short h0 = f32_bf16_rne(v.x), h1 = f32_bf16_rne(v.y);
        unsigned short h2 = f32_bf16_rne(v.z), h3 = f32_bf16_rne(v.w);
        unsigned short l0 = f32_bf16_rne(v.x - bf16_as_f32(h0));
        unsigned short l1 = f32_bf16_rne(v.y - bf16_as_f32(h1));
        unsigned short l2 = f32_bf16_rne(v.z - bf16_as_f32(h2));
        unsigned short l3 = f32_bf16_rne(v.w - bf16_as_f32(h3));
        dsth[i] = make_uint2((unsigned)h0 | ((unsigned)h1 << 16),
                             (unsigned)h2 | ((unsigned)h3 << 16));
        dstl[i] = make_uint2((unsigned)l0 | ((unsigned)l1 << 16),
                             (unsigned)l2 | ((unsigned)l3 << 16));
    }
    __syncthreads();

    const int lane = threadIdx.x & 31;
    const int wave = threadIdx.x >> 5;
    const int col  = lane & 15;   // N-col within tile / M-row for A frag
    const int half = lane >> 4;   // lane half selects K-halves per ISA layout

    const unsigned short* aHiRow = a_hi + (size_t)(wave * 16 + col) * ENC2_;
    const unsigned short* aLoRow = a_lo + (size_t)(wave * 16 + col) * ENC2_;

    float acc[8] = {0.f, 0.f, 0.f, 0.f, 0.f, 0.f, 0.f, 0.f};

    union Frag { uint4 q[2]; v16bf v; };

    for (int n0 = 0; n0 < ATT_; n0 += 32) {      // two 16-wide N-tiles / iter
        const int nA = n0 + col;
        const int nB = n0 + 16 + col;
        const unsigned short* bHiA = whi + (size_t)nA * ENC2_;
        const unsigned short* bLoA = wlo + (size_t)nA * ENC2_;
        const unsigned short* bHiB = whi + (size_t)nB * ENC2_;
        const unsigned short* bLoB = wlo + (size_t)nB * ENC2_;

        v8f chhA = {}, cloA = {}, chhB = {}, cloB = {};

        for (int kb = 0; kb < ENC2_; kb += KTILE) {
            const int o1 = kb + half * 8;        // K[half*8 .. +8)
            const int o2 = kb + 16 + half * 8;   // K[16+half*8 .. +8)
            Frag ah, al, bhA, blA, bhB, blB;
            ah.q[0]  = *(const uint4*)(aHiRow + o1);
            ah.q[1]  = *(const uint4*)(aHiRow + o2);
            al.q[0]  = *(const uint4*)(aLoRow + o1);
            al.q[1]  = *(const uint4*)(aLoRow + o2);
            bhA.q[0] = *(const uint4*)(bHiA + o1);
            bhA.q[1] = *(const uint4*)(bHiA + o2);
            blA.q[0] = *(const uint4*)(bLoA + o1);
            blA.q[1] = *(const uint4*)(bLoA + o2);
            bhB.q[0] = *(const uint4*)(bHiB + o1);
            bhB.q[1] = *(const uint4*)(bHiB + o2);
            blB.q[0] = *(const uint4*)(bLoB + o1);
            blB.q[1] = *(const uint4*)(bLoB + o2);
            // 4 independent accumulation chains, 6 WMMAs per K-step
            chhA = __builtin_amdgcn_wmma_f32_16x16x32_bf16(false, ah.v, false, bhA.v,
                                                           (short)0, chhA, false, false);
            chhB = __builtin_amdgcn_wmma_f32_16x16x32_bf16(false, ah.v, false, bhB.v,
                                                           (short)0, chhB, false, false);
            cloA = __builtin_amdgcn_wmma_f32_16x16x32_bf16(false, al.v, false, bhA.v,
                                                           (short)0, cloA, false, false);
            cloB = __builtin_amdgcn_wmma_f32_16x16x32_bf16(false, al.v, false, bhB.v,
                                                           (short)0, cloB, false, false);
            cloA = __builtin_amdgcn_wmma_f32_16x16x32_bf16(false, ah.v, false, blA.v,
                                                           (short)0, cloA, false, false);
            cloB = __builtin_amdgcn_wmma_f32_16x16x32_bf16(false, ah.v, false, blB.v,
                                                           (short)0, cloB, false, false);
        }

        const float dpA = dproj[(size_t)b * ATT_ + nA];
        const float weA = w_e[nA];
        const float dpB = dproj[(size_t)b * ATT_ + nB];
        const float weB = w_e[nB];
#pragma unroll
        for (int r = 0; r < 8; ++r) {
            acc[r] += tanhf(chhA[r] + cloA[r] + dpA) * weA;
            acc[r] += tanhf(chhB[r] + cloB[r] + dpB) * weB;
        }
    }

    // C layout: VGPR r, lanes 0-15 -> row r, lanes 16-31 -> row r+8.
    // Sum each row's partials across the 16 lanes of its half.
#pragma unroll
    for (int r = 0; r < 8; ++r) {
        float v = acc[r];
        v += __shfl_xor(v, 1);
        v += __shfl_xor(v, 2);
        v += __shfl_xor(v, 4);
        v += __shfl_xor(v, 8);
        acc[r] = v;
    }
    if (col == 0) {
        const float be = b_e_p[0];
#pragma unroll
        for (int r = 0; r < 8; ++r) {
            int s = s0 + wave * 16 + half * 8 + r;
            size_t idx = (size_t)b * S_ + s;
            float e = acc[r] + be;
            energy[idx] = mask[idx] ? e : -__builtin_inff();
        }
    }
}

// ---------------------------------------------------------------------------
// Row softmax over S=2048 per batch.
// ---------------------------------------------------------------------------
__global__ __launch_bounds__(256) void softmax_kernel(
    const float* __restrict__ energy, float* __restrict__ wts) {
    __shared__ float red[256];
    const int b = blockIdx.x;
    const int tid = threadIdx.x;
    const float* e = energy + (size_t)b * S_;

    float m = -__builtin_inff();
    for (int i = tid; i < S_; i += 256) m = fmaxf(m, e[i]);
    red[tid] = m; __syncthreads();
    for (int o = 128; o > 0; o >>= 1) {
        if (tid < o) red[tid] = fmaxf(red[tid], red[tid + o]);
        __syncthreads();
    }
    m = red[0]; __syncthreads();

    float s = 0.f;
    for (int i = tid; i < S_; i += 256) s += expf(e[i] - m);
    red[tid] = s; __syncthreads();
    for (int o = 128; o > 0; o >>= 1) {
        if (tid < o) red[tid] += red[tid + o];
        __syncthreads();
    }
    s = red[0];
    const float inv = (s > 0.f) ? 1.f / s : 0.f;

    float* w = wts + (size_t)b * S_;
    for (int i = tid; i < S_; i += 256) w[i] = expf(e[i] - m) * inv;
}

// ---------------------------------------------------------------------------
// context[b][e] = sum_s weights[b][s] * enc[b][s][e]   (coalesced stream)
// ---------------------------------------------------------------------------
__global__ __launch_bounds__(128) void context_kernel(
    const float* __restrict__ enc, const float* __restrict__ wts,
    float* __restrict__ ctx) {
    const int b = blockIdx.x;
    const int e = blockIdx.y * 128 + threadIdx.x;
    const float* w = wts + (size_t)b * S_;
    const float* p = enc + (size_t)b * S_ * ENC2_ + e;
    float acc = 0.f;
    for (int s = 0; s < S_; ++s)
        acc = fmaf(w[s], p[(size_t)s * ENC2_], acc);
    ctx[(size_t)b * ENC2_ + e] = acc;
}

// ---------------------------------------------------------------------------
extern "C" void kernel_launch(void* const* d_in, const int* in_sizes, int n_in,
                              void* d_out, int out_size, void* d_ws, size_t ws_size,
                              hipStream_t stream) {
    const float* dec_hidden       = (const float*)d_in[0];
    const float* enc              = (const float*)d_in[1];
    const unsigned char* mask     = (const unsigned char*)d_in[2];
    const float* Wenc             = (const float*)d_in[3];
    const float* benc             = (const float*)d_in[4];
    const float* Wdec             = (const float*)d_in[5];
    const float* bdec             = (const float*)d_in[6];
    const float* w_e              = (const float*)d_in[7];
    const float* b_e              = (const float*)d_in[8];

    float* ctx = (float*)d_out;                      // (B, ENC2)
    float* wts = (float*)d_out + (size_t)B_ * ENC2_; // (B, S)

    char* ws = (char*)d_ws;
    unsigned short* whi = (unsigned short*)(ws);                           // 2 MB
    unsigned short* wlo = (unsigned short*)(ws + (size_t)2 * 1024 * 1024); // 2 MB
    float* dproj  = (float*)(ws + (size_t)4 * 1024 * 1024);                // 128 KB
    float* energy = (float*)(ws + (size_t)4 * 1024 * 1024 + 128 * 1024);   // 256 KB

    wprep_kernel<<<(ENC2_ * ATT_) / 256, 256, 0, stream>>>(Wenc, whi, wlo);
    dproj_kernel<<<B_, 256, 0, stream>>>(dec_hidden, Wdec, bdec, benc, dproj);
    energy_kernel<<<B_ * (S_ / ROWS), 128, 0, stream>>>(
        enc, whi, wlo, dproj, w_e, b_e, mask, energy);
    softmax_kernel<<<B_, 256, 0, stream>>>(energy, wts);
    context_kernel<<<dim3(B_, ENC2_ / 128), 128, 0, stream>>>(enc, wts, ctx);
}
